// MEGNetRL_19413252178604
// MI455X (gfx1250) — matrix operand order repaired and checked by reading, not compiled, not run on hardware
//
#include <hip/hip_runtime.h>
#include <hip/hip_bf16.h>
#include <math.h>

typedef __attribute__((ext_vector_type(16))) _Float16 v16h;
typedef __attribute__((ext_vector_type(8)))  float    v8f;
typedef _Float16 half_t;

#define NN 50000
#define NE 800000
#define G_RED 1024

// ---------------- scalar helpers ----------------
// Fast branchless softplus2: log(1+exp(x)) - log(2) using hardware v_exp_f32 / v_log_f32.
// For x>20, log1p(exp(x)) == x to f32 precision; guard via cndmask (no branch).
__device__ __forceinline__ float sp2f(float x) {
  float e  = __expf(x);
  float sp = __logf(1.0f + e) - 0.69314718055994530942f;
  float bg = x - 0.69314718055994530942f;
  return (x > 20.0f) ? bg : sp;
}
// Precise versions for the tiny scalar kernels (1 workgroup, negligible runtime).
__device__ __forceinline__ float sp2f_precise(float x) {
  float sp = (x > 20.0f) ? x : log1pf(expf(x));
  return sp - 0.69314718055994530942f;
}
__device__ __forceinline__ float sigf(float x) { return 1.0f / (1.0f + expf(-x)); }

// ---------------- WMMA tile helpers (wave32) ----------------
// A fragment layout (ISA 7.12.2, 16-bit A 16x32): lane half=lane>>4, m=lane&15,
//   vgpr v, elem p: k = kbase + (v<4?0:16) + 8*half + 2*(v&3) + p
// B fragment (32x16): n=lane&15, half=lane>>4,
//   vgpr v, elem p: k = kbase + (v<4?0:16) + 8*half + 2*(v&3) + p
// C/D: vgpr r -> M = r + 8*half, N = lane&15
template <int KP, int DOUT>
__device__ __forceinline__ void wave_gemm16(const half_t* __restrict__ A,   // wave's 16-row block, stride KP+4
                                            const half_t* __restrict__ Wt,  // [DOUT][KP+4] (k-contig per out col)
                                            const float* __restrict__ bias,
                                            int lane, v8f acc[DOUT / 16]) {
  const int half = lane >> 4, nl = lane & 15, m = lane & 15;
  constexpr int NK = KP / 32;
  v16h af[NK];
#pragma unroll
  for (int kb = 0; kb < NK; ++kb) {
    const half_t* ar = A + m * (KP + 4) + kb * 32 + half * 8;
#pragma unroll
    for (int v = 0; v < 8; ++v) {
      int k = ((v < 4) ? 0 : 16) + (v & 3) * 2;
      af[kb][2 * v]     = ar[k];
      af[kb][2 * v + 1] = ar[k + 1];
    }
  }
#pragma unroll
  for (int nt = 0; nt < DOUT / 16; ++nt) {
    float b = bias[nt * 16 + nl];
    v8f c = {b, b, b, b, b, b, b, b};
#pragma unroll
    for (int kb = 0; kb < NK; ++kb) {
      const half_t* wr = Wt + (nt * 16 + nl) * (KP + 4) + kb * 32 + half * 8;
      v16h bf;
#pragma unroll
      for (int v = 0; v < 8; ++v) {
        int k = ((v < 4) ? 0 : 16) + (v & 3) * 2;
        bf[2 * v]     = wr[k];
        bf[2 * v + 1] = wr[k + 1];
      }
      c = __builtin_amdgcn_wmma_f32_16x16x32_f16(false, af[kb], false, bf, (short)0, c, false, false);
    }
    acc[nt] = c;
  }
}

template <int KP, int DOUT>
__device__ __forceinline__ void stage_w(const float* __restrict__ Wg, const float* __restrict__ Bg,
                                        int Kreal, half_t* Wt, float* bias) {
  const int tid = threadIdx.x, NT = blockDim.x;
  for (int idx = tid; idx < KP * DOUT; idx += NT) {
    int k = idx / DOUT, n = idx - k * DOUT;
    float w = (k < Kreal) ? Wg[k * DOUT + n] : 0.0f;
    Wt[n * (KP + 4) + k] = (half_t)w;
  }
  for (int n = tid; n < DOUT; n += NT) bias[n] = Bg[n];
}

template <int DOUT>
__device__ __forceinline__ void store_act(half_t* Aout, int ostr, int lane, const v8f* acc) {
  const int half = lane >> 4, nl = lane & 15;
#pragma unroll
  for (int nt = 0; nt < DOUT / 16; ++nt)
#pragma unroll
    for (int r = 0; r < 8; ++r)
      Aout[(r + 8 * half) * ostr + nt * 16 + nl] = (half_t)sp2f(acc[nt][r]);
}

// ---------------- small utility kernels ----------------
__global__ void mg_fill(float* p, long n, float v) {
  for (long i = blockIdx.x * (long)blockDim.x + threadIdx.x; i < n; i += (long)gridDim.x * blockDim.x)
    p[i] = v;
}
__global__ void mg_cvt(const float* __restrict__ x, half_t* __restrict__ y, long n) {
  for (long i = blockIdx.x * (long)blockDim.x + threadIdx.x; i < n; i += (long)gridDim.x * blockDim.x)
    y[i] = (half_t)x[i];
}
__global__ void mg_copyf(const float* __restrict__ x, float* __restrict__ y, int n) {
  if (threadIdx.x < n) y[threadIdx.x] = x[threadIdx.x];
}
__global__ void mg_deg(const int* __restrict__ dst, float* __restrict__ deg, int E) {
  for (long i = blockIdx.x * (long)blockDim.x + threadIdx.x; i < E; i += (long)gridDim.x * blockDim.x)
    atomicAdd(&deg[dst[i]], 1.0f);
}
__global__ void mg_focus(const int* __restrict__ nf, int* __restrict__ focus, int N) {
  for (long i = blockIdx.x * (long)blockDim.x + threadIdx.x; i < N; i += (long)gridDim.x * blockDim.x)
    if (nf[i * 90 + 89] != 0) *focus = (int)i;
}

// single-block scalar MLP layer: out[j] = act(b[j] + sum_k in[k]*W[k*dout+j])
__device__ __forceinline__ void blk_layer(const float* in, int din, const float* W, const float* B,
                                          float* out, int dout, bool act) {
  for (int j = threadIdx.x; j < dout; j += blockDim.x) {
    float s = B[j];
    for (int k = 0; k < din; ++k) s += in[k] * W[k * dout + j];
    out[j] = act ? sp2f_precise(s) : s;
  }
  __syncthreads();
}

// ---------------- encoders ----------------
__global__ __launch_bounds__(128) void mg_node_init(const int* __restrict__ nf,
                                                    const float* __restrict__ emb,
                                                    const float* W0, const float* B0,
                                                    const float* W1, const float* B1,
                                                    float* __restrict__ v_cur, int N) {
  __shared__ half_t A0[64 * 36];
  __shared__ half_t A1[64 * 68];
  __shared__ half_t Wt[64 * 68];
  __shared__ float bias[64];
  const int tid = threadIdx.x, base = blockIdx.x * 64;
  if (tid < 64) {
    int g = base + tid;
    if (g < N) {
      int best = -1, idx = 0;
      for (int t = 0; t < 89; ++t) {
        int vv = nf[(long)g * 90 + t];
        if (vv > best) { best = vv; idx = t; }
      }
      for (int k = 0; k < 16; ++k) A0[tid * 36 + k] = (half_t)emb[idx * 16 + k];
      for (int k = 16; k < 32; ++k) A0[tid * 36 + k] = (half_t)0.0f;
    } else {
      for (int k = 0; k < 32; ++k) A0[tid * 36 + k] = (half_t)0.0f;
    }
  }
  const int wave = tid >> 5, lane = tid & 31;
  __syncthreads();
  stage_w<32, 64>(W0, B0, 16, Wt, bias);
  __syncthreads();
  { v8f a[4]; wave_gemm16<32, 64>(A0 + wave * 16 * 36, Wt, bias, lane, a); store_act<64>(A1 + wave * 16 * 68, 68, lane, a); }
  __syncthreads();
  stage_w<64, 32>(W1, B1, 64, Wt, bias);
  __syncthreads();
  {
    v8f a[2];
    wave_gemm16<64, 32>(A1 + wave * 16 * 68, Wt, bias, lane, a);
    const int half = lane >> 4, nl = lane & 15;
#pragma unroll
    for (int nt = 0; nt < 2; ++nt)
#pragma unroll
      for (int r = 0; r < 8; ++r) {
        int g = base + wave * 16 + r + 8 * half;
        if (g < N) v_cur[(long)g * 32 + nt * 16 + nl] = sp2f(a[nt][r]);
      }
  }
}

__global__ __launch_bounds__(128) void mg_edge_init(const float* __restrict__ dist,
                                                    const float* W0, const float* B0,
                                                    const float* W1, const float* B1,
                                                    float* __restrict__ e_cur, int E) {
  __shared__ half_t A0[64 * 132];
  __shared__ half_t A1[64 * 68];
  __shared__ half_t Wt[64 * 132];
  __shared__ float bias[64];
  __shared__ float sd[64];
  const int tid = threadIdx.x, base = blockIdx.x * 64;
  if (tid < 64) { int g = base + tid; sd[tid] = (g < E) ? dist[g] : 0.0f; }
  __syncthreads();
  const float step = 4.0f / 99.0f;
  for (int j = 0; j < 64; ++j) {
    int c = tid;  // 128 cols
    float val = 0.0f;
    if ((base + j) < E && c < 100) {
      float t = (sd[j] - (float)c * step) * 2.0f;  // /WIDTH(0.5)
      val = __expf(-t * t);
    }
    A0[j * 132 + c] = (half_t)val;
  }
  const int wave = tid >> 5, lane = tid & 31;
  __syncthreads();
  stage_w<128, 64>(W0, B0, 100, Wt, bias);
  __syncthreads();
  { v8f a[4]; wave_gemm16<128, 64>(A0 + wave * 16 * 132, Wt, bias, lane, a); store_act<64>(A1 + wave * 16 * 68, 68, lane, a); }
  __syncthreads();
  stage_w<64, 32>(W1, B1, 64, Wt, bias);
  __syncthreads();
  {
    v8f a[2];
    wave_gemm16<64, 32>(A1 + wave * 16 * 68, Wt, bias, lane, a);
    const int half = lane >> 4, nl = lane & 15;
#pragma unroll
    for (int nt = 0; nt < 2; ++nt)
#pragma unroll
      for (int r = 0; r < 8; ++r) {
        int g = base + wave * 16 + r + 8 * half;
        if (g < E) e_cur[(long)g * 32 + nt * 16 + nl] = sp2f(a[nt][r]);
      }
  }
}

__global__ void mg_state_enc(const float* __restrict__ sf, const float* __restrict__ semb,
                             const int* __restrict__ focus,
                             const float* W0, const float* B0, const float* W1, const float* B1,
                             float* __restrict__ u_out) {
  __shared__ float z0[16], z1[64], z2[32];
  int tid = threadIdx.x;
  if (tid < 8) { z0[tid] = sf[tid]; z0[8 + tid] = semb[(*focus) * 8 + tid]; }
  __syncthreads();
  blk_layer(z0, 16, W0, B0, z1, 64, true);
  blk_layer(z1, 64, W1, B1, z2, 32, true);
  if (tid < 32) u_out[tid] = z2[tid];
}

// ---------------- dense (pre-conv) ----------------
__global__ __launch_bounds__(128) void mg_dense_rows(const float* __restrict__ X,
                                                     const float* W0, const float* B0,
                                                     const float* W1, const float* B1,
                                                     half_t* __restrict__ XD, int M) {
  __shared__ half_t A0[64 * 36];
  __shared__ half_t A1[64 * 68];
  __shared__ half_t Wt[64 * 68];
  __shared__ float bias[64];
  const int tid = threadIdx.x, base = blockIdx.x * 64;
  for (int j = 0; j < 64; ++j) {
    if (tid < 32) {
      int g = base + j;
      A0[j * 36 + tid] = (half_t)((g < M) ? X[(long)g * 32 + tid] : 0.0f);
    }
  }
  const int wave = tid >> 5, lane = tid & 31;
  __syncthreads();
  stage_w<32, 64>(W0, B0, 32, Wt, bias);
  __syncthreads();
  { v8f a[4]; wave_gemm16<32, 64>(A0 + wave * 16 * 36, Wt, bias, lane, a); store_act<64>(A1 + wave * 16 * 68, 68, lane, a); }
  __syncthreads();
  stage_w<64, 32>(W1, B1, 64, Wt, bias);
  __syncthreads();
  {
    v8f a[2];
    wave_gemm16<64, 32>(A1 + wave * 16 * 68, Wt, bias, lane, a);
    const int half = lane >> 4, nl = lane & 15;
#pragma unroll
    for (int nt = 0; nt < 2; ++nt)
#pragma unroll
      for (int r = 0; r < 8; ++r) {
        int g = base + wave * 16 + r + 8 * half;
        if (g < M) XD[(long)g * 32 + nt * 16 + nl] = (half_t)sp2f(a[nt][r]);
      }
  }
}

__global__ void mg_dense_state(const float* __restrict__ u_in,
                               const float* W0, const float* B0, const float* W1, const float* B1,
                               float* __restrict__ uD) {
  __shared__ float z0[32], z1[64], z2[32];
  int tid = threadIdx.x;
  if (tid < 32) z0[tid] = u_in[tid];
  __syncthreads();
  blk_layer(z0, 32, W0, B0, z1, 64, true);
  blk_layer(z1, 64, W1, B1, z2, 32, true);
  if (tid < 32) uD[tid] = z2[tid];
}

// ---------------- conv: edge update ----------------
__global__ __launch_bounds__(128) void mg_edge_update(const int* __restrict__ src, const int* __restrict__ dst,
                                                      const half_t* __restrict__ eD, const half_t* __restrict__ vD,
                                                      const float* __restrict__ uD,
                                                      const float* W0, const float* B0,
                                                      const float* W1, const float* B1,
                                                      const float* W2, const float* B2,
                                                      float* __restrict__ e_cur, float* __restrict__ e_sum,
                                                      float* __restrict__ eg_sum, int E) {
  __shared__ half_t A0[64 * 132];
  __shared__ half_t A1[64 * 68];
  __shared__ half_t Wt[64 * 132];
  __shared__ float bias[64];
  __shared__ int ssrc[64], sdst[64];
  __shared__ float us[32], egl[32];
  const int tid = threadIdx.x, base = blockIdx.x * 64;
  if (tid < 64) {
    int g = base + tid;
    ssrc[tid] = (g < E) ? src[g] : 0;
    sdst[tid] = (g < E) ? dst[g] : 0;
  }
  if (tid < 32) { us[tid] = uD[tid]; egl[tid] = 0.0f; }
  __syncthreads();
  for (int j = 0; j < 64; ++j) {
    int g = base + j, c = tid;
    half_t v;
    if (g >= E)      v = (half_t)0.0f;
    else if (c < 32) v = vD[(long)ssrc[j] * 32 + c];
    else if (c < 64) v = vD[(long)sdst[j] * 32 + (c - 32)];
    else if (c < 96) v = eD[(long)g * 32 + (c - 64)];
    else             v = (half_t)us[c - 96];
    A0[j * 132 + c] = v;
  }
  const int wave = tid >> 5, lane = tid & 31;
  __syncthreads();
  stage_w<128, 64>(W0, B0, 128, Wt, bias);
  __syncthreads();
  { v8f a[4]; wave_gemm16<128, 64>(A0 + wave * 16 * 132, Wt, bias, lane, a); store_act<64>(A1 + wave * 16 * 68, 68, lane, a); }
  __syncthreads();
  stage_w<64, 64>(W1, B1, 64, Wt, bias);
  __syncthreads();
  { v8f a[4]; wave_gemm16<64, 64>(A1 + wave * 16 * 68, Wt, bias, lane, a); store_act<64>(A1 + wave * 16 * 68, 68, lane, a); }
  __syncthreads();
  stage_w<64, 32>(W2, B2, 64, Wt, bias);
  __syncthreads();
  {
    v8f a[2];
    wave_gemm16<64, 32>(A1 + wave * 16 * 68, Wt, bias, lane, a);
    const int half = lane >> 4, nl = lane & 15;
#pragma unroll
    for (int nt = 0; nt < 2; ++nt)
#pragma unroll
      for (int r = 0; r < 8; ++r) {
        int rloc = wave * 16 + r + 8 * half;
        int g = base + rloc;
        if (g < E) {
          float enew = sp2f(a[nt][r]);
          int col = nt * 16 + nl;
          e_cur[(long)g * 32 + col] += enew;                       // residual e = e_new + e0
          atomicAdd(&e_sum[(long)sdst[rloc] * 32 + col], enew);    // segment sum for e_mean
          atomicAdd(&egl[col], enew);                              // for global mean
        }
      }
  }
  __syncthreads();
  if (tid < 32) atomicAdd(&eg_sum[tid], egl[tid]);
}

// ---------------- conv: node update ----------------
__global__ __launch_bounds__(128) void mg_node_update(const half_t* __restrict__ vD,
                                                      const float* __restrict__ e_sum, const float* __restrict__ degv,
                                                      const float* __restrict__ uD,
                                                      const float* W0, const float* B0,
                                                      const float* W1, const float* B1,
                                                      const float* W2, const float* B2,
                                                      float* __restrict__ v_cur, float* __restrict__ vg_sum, int N) {
  __shared__ half_t A0[64 * 100];
  __shared__ half_t A1[64 * 68];
  __shared__ half_t Wt[64 * 100];
  __shared__ float bias[64];
  __shared__ float sinv[64], us[32], vgl[32];
  const int tid = threadIdx.x, base = blockIdx.x * 64;
  if (tid < 64) {
    int g = base + tid;
    float d = (g < N) ? degv[g] : 1.0f;
    sinv[tid] = 1.0f / fmaxf(d, 1.0f);
  }
  if (tid < 32) { us[tid] = uD[tid]; vgl[tid] = 0.0f; }
  __syncthreads();
  for (int j = 0; j < 64; ++j) {
    if (tid < 96) {
      int g = base + j, c = tid;
      float val = 0.0f;
      if (g < N) {
        if (c < 32)      val = (float)vD[(long)g * 32 + c];
        else if (c < 64) val = e_sum[(long)g * 32 + (c - 32)] * sinv[j];
        else             val = us[c - 64];
      }
      A0[j * 100 + c] = (half_t)val;
    }
  }
  const int wave = tid >> 5, lane = tid & 31;
  __syncthreads();
  stage_w<96, 64>(W0, B0, 96, Wt, bias);
  __syncthreads();
  { v8f a[4]; wave_gemm16<96, 64>(A0 + wave * 16 * 100, Wt, bias, lane, a); store_act<64>(A1 + wave * 16 * 68, 68, lane, a); }
  __syncthreads();
  stage_w<64, 64>(W1, B1, 64, Wt, bias);
  __syncthreads();
  { v8f a[4]; wave_gemm16<64, 64>(A1 + wave * 16 * 68, Wt, bias, lane, a); store_act<64>(A1 + wave * 16 * 68, 68, lane, a); }
  __syncthreads();
  stage_w<64, 32>(W2, B2, 64, Wt, bias);
  __syncthreads();
  {
    v8f a[2];
    wave_gemm16<64, 32>(A1 + wave * 16 * 68, Wt, bias, lane, a);
    const int half = lane >> 4, nl = lane & 15;
#pragma unroll
    for (int nt = 0; nt < 2; ++nt)
#pragma unroll
      for (int r = 0; r < 8; ++r) {
        int g = base + wave * 16 + r + 8 * half;
        if (g < N) {
          float vnew = sp2f(a[nt][r]);
          int col = nt * 16 + nl;
          v_cur[(long)g * 32 + col] += vnew;     // residual v = v_new + v0
          atomicAdd(&vgl[col], vnew);
        }
      }
  }
  __syncthreads();
  if (tid < 32) atomicAdd(&vg_sum[tid], vgl[tid]);
}

__global__ void mg_state_update(const float* __restrict__ eg, const float* __restrict__ vg,
                                const float* __restrict__ uD,
                                const float* W0, const float* B0, const float* W1, const float* B1,
                                const float* W2, const float* B2,
                                float* __restrict__ u_cur, float invE, float invN) {
  __shared__ float z0[96], z1[64], z2[64], z3[32];
  int tid = threadIdx.x;
  if (tid < 32) { z0[tid] = eg[tid] * invE; z0[32 + tid] = vg[tid] * invN; z0[64 + tid] = uD[tid]; }
  __syncthreads();
  blk_layer(z0, 96, W0, B0, z1, 64, true);
  blk_layer(z1, 64, W1, B1, z2, 64, true);
  blk_layer(z2, 64, W2, B2, z3, 32, true);
  if (tid < 32) u_cur[tid] += z3[tid];           // residual u = u_new + u0
}

// ---------------- set2set ----------------
__global__ void mg_lstm(const float* __restrict__ q, float* __restrict__ h, float* __restrict__ c,
                        const float* __restrict__ Wih, const float* __restrict__ Whh, const float* __restrict__ b) {
  __shared__ float z[128], qs[64], hs[32], cs[32];
  int tid = threadIdx.x;
  if (tid < 64) qs[tid] = q[tid];
  if (tid < 32) { hs[tid] = h[tid]; cs[tid] = c[tid]; }
  __syncthreads();
  {
    float s = b[tid];
    for (int k = 0; k < 64; ++k) s += qs[k] * Wih[k * 128 + tid];
    for (int k = 0; k < 32; ++k) s += hs[k] * Whh[k * 128 + tid];
    z[tid] = s;
  }
  __syncthreads();
  if (tid < 32) {
    float cn = sigf(z[32 + tid]) * cs[tid] + sigf(z[tid]) * tanhf(z[64 + tid]);
    float hn = sigf(z[96 + tid]) * tanhf(cn);
    c[tid] = cn;
    h[tid] = hn;
  }
}

__global__ __launch_bounds__(256) void mg_logits(const float* __restrict__ feat, int M,
                                                 const float* __restrict__ h,
                                                 float* __restrict__ logits, float* __restrict__ pmax) {
  __shared__ float hs[32];
  __shared__ float red[256];
  int tid = threadIdx.x;
  if (tid < 32) hs[tid] = h[tid];
  __syncthreads();
  float lm = -3.0e38f;
  for (long row = blockIdx.x * 256L + tid; row < M; row += (long)gridDim.x * 256L) {
    const float* f = feat + row * 32;
    float s = 0.0f;
    for (int c = 0; c < 32; ++c) s += f[c] * hs[c];
    logits[row] = s;
    lm = fmaxf(lm, s);
  }
  red[tid] = lm;
  __syncthreads();
  for (int s = 128; s > 0; s >>= 1) {
    if (tid < s) red[tid] = fmaxf(red[tid], red[tid + s]);
    __syncthreads();
  }
  if (tid == 0) pmax[blockIdx.x] = red[0];
}

__global__ void mg_maxred(const float* __restrict__ pmax, int G, float* __restrict__ gm) {
  __shared__ float red[256];
  int tid = threadIdx.x;
  float m = -3.0e38f;
  for (int g = tid; g < G; g += 256) m = fmaxf(m, pmax[g]);
  red[tid] = m;
  __syncthreads();
  for (int s = 128; s > 0; s >>= 1) {
    if (tid < s) red[tid] = fmaxf(red[tid], red[tid + s]);
    __syncthreads();
  }
  if (tid == 0) gm[0] = red[0];
}

__global__ __launch_bounds__(256) void mg_expsum(const float* __restrict__ feat, int M,
                                                 const float* __restrict__ logits, const float* __restrict__ gm,
                                                 float* __restrict__ psum, float* __restrict__ pr) {
  __shared__ float racc[32];
  __shared__ float sacc;
  int tid = threadIdx.x;
  if (tid < 32) racc[tid] = 0.0f;
  if (tid == 0) sacc = 0.0f;
  __syncthreads();
  float m = gm[0], sl = 0.0f, rl[32];
#pragma unroll
  for (int c = 0; c < 32; ++c) rl[c] = 0.0f;
  for (long row = blockIdx.x * 256L + tid; row < M; row += (long)gridDim.x * 256L) {
    float w = __expf(logits[row] - m);   // w in (0,1], fast exp is plenty
    sl += w;
    const float* f = feat + row * 32;
#pragma unroll
    for (int c = 0; c < 32; ++c) rl[c] += w * f[c];
  }
#pragma unroll
  for (int c = 0; c < 32; ++c) atomicAdd(&racc[c], rl[c]);
  atomicAdd(&sacc, sl);
  __syncthreads();
  if (tid < 32) pr[blockIdx.x * 32 + tid] = racc[tid];
  if (tid == 0) psum[blockIdx.x] = sacc;
}

__global__ void mg_s2s_final(const float* __restrict__ psum, const float* __restrict__ pr, int G,
                             const float* __restrict__ h, float* __restrict__ q, float* __restrict__ outvec) {
  __shared__ float racc[32];
  __shared__ float sacc;
  int tid = threadIdx.x;
  if (tid < 32) racc[tid] = 0.0f;
  if (tid == 0) sacc = 0.0f;
  __syncthreads();
  float sl = 0.0f, rl[32];
#pragma unroll
  for (int c = 0; c < 32; ++c) rl[c] = 0.0f;
  for (int g = tid; g < G; g += blockDim.x) {
    sl += psum[g];
    const float* p = pr + g * 32;
#pragma unroll
    for (int c = 0; c < 32; ++c) rl[c] += p[c];
  }
  atomicAdd(&sacc, sl);
#pragma unroll
  for (int c = 0; c < 32; ++c) atomicAdd(&racc[c], rl[c]);
  __syncthreads();
  if (tid < 32) {
    float r = racc[tid] / sacc;
    q[tid] = h[tid];
    q[32 + tid] = r;
    outvec[tid] = h[tid];
    outvec[32 + tid] = r;
  }
}

__global__ void mg_out(const float* __restrict__ nvec, const float* __restrict__ evec,
                       const float* __restrict__ u,
                       const float* W0, const float* B0, const float* W1, const float* B1,
                       const float* W2, const float* B2, float* __restrict__ out) {
  __shared__ float z0[160], z1[32], z2[16];
  int tid = threadIdx.x;
  if (tid < 64) { z0[tid] = nvec[tid]; z0[64 + tid] = evec[tid]; }
  if (tid < 32) z0[128 + tid] = u[tid];
  __syncthreads();
  blk_layer(z0, 160, W0, B0, z1, 32, true);
  blk_layer(z1, 32, W1, B1, z2, 16, true);
  blk_layer(z2, 16, W2, B2, out, 88, false);     // activate_last=False
}

// ---------------- host ----------------
extern "C" void kernel_launch(void* const* d_in, const int* in_sizes, int n_in,
                              void* d_out, int out_size, void* d_ws, size_t ws_size,
                              hipStream_t stream) {
  (void)in_sizes; (void)n_in; (void)out_size; (void)ws_size;
  const int E = NE, N = NN;
  int ix = 0;
  auto nf_ = [&]() { return (const float*)d_in[ix++]; };

  // params (insertion order of make_params(), recursive; None leaves skipped)
  const float* node_emb  = nf_();
  const float* state_emb = nf_();
  const float *eeW0 = nf_(), *eeB0 = nf_(), *eeW1 = nf_(), *eeB1 = nf_();  // edge_enc
  const float *neW0 = nf_(), *neB0 = nf_(), *neW1 = nf_(), *neB1 = nf_();  // node_enc
  const float *scW0 = nf_(), *scB0 = nf_(), *scW1 = nf_(), *scB1 = nf_();  // state_enc
  const float *deW0[3] = {0}, *deB0[3] = {0}, *deW1[3] = {0}, *deB1[3] = {0};
  const float *dnW0[3] = {0}, *dnB0[3] = {0}, *dnW1[3] = {0}, *dnB1[3] = {0};
  const float *dsW0[3] = {0}, *dsB0[3] = {0}, *dsW1[3] = {0}, *dsB1[3] = {0};
  const float *edW[3][3], *edB[3][3], *ndW[3][3], *ndB[3][3], *stW[3][3], *stB[3][3];
  for (int b = 0; b < 3; ++b) {
    if (b > 0) {
      deW0[b] = nf_(); deB0[b] = nf_(); deW1[b] = nf_(); deB1[b] = nf_();
      dnW0[b] = nf_(); dnB0[b] = nf_(); dnW1[b] = nf_(); dnB1[b] = nf_();
      dsW0[b] = nf_(); dsB0[b] = nf_(); dsW1[b] = nf_(); dsB1[b] = nf_();
    }
    for (int l = 0; l < 3; ++l) { edW[b][l] = nf_(); edB[b][l] = nf_(); }
    for (int l = 0; l < 3; ++l) { ndW[b][l] = nf_(); ndB[b][l] = nf_(); }
    for (int l = 0; l < 3; ++l) { stW[b][l] = nf_(); stB[b][l] = nf_(); }
  }
  const float *snWih = nf_(), *snWhh = nf_(), *snB = nf_();  // s2s_node
  const float *szWih = nf_(), *szWhh = nf_(), *szB = nf_();  // s2s_edge
  const float *oW0 = nf_(), *oB0 = nf_(), *oW1 = nf_(), *oB1 = nf_(), *oW2 = nf_(), *oB2 = nf_();
  const float* edge_dist = nf_();
  const int* node_feat = (const int*)d_in[ix++];
  const float* state_feat = nf_();
  const int* srcp = (const int*)d_in[ix++];
  const int* dstp = (const int*)d_in[ix++];

  // workspace carve (256B aligned)
  char* wp = (char*)d_ws;
  auto alloc = [&](size_t bytes) { void* p = (void*)wp; wp += (bytes + 255) & ~(size_t)255; return p; };
  float*  e_cur = (float*)alloc((size_t)E * 32 * 4);
  half_t* eD    = (half_t*)alloc((size_t)E * 32 * 2);
  float*  v_cur = (float*)alloc((size_t)N * 32 * 4);
  half_t* vD    = (half_t*)alloc((size_t)N * 32 * 2);
  float*  e_sum = (float*)alloc(((size_t)N * 32 + 64) * 4);   // e_sum | eg_sum(32) | vg_sum(32)
  float*  eg_sum = e_sum + (size_t)N * 32;
  float*  vg_sum = eg_sum + 32;
  float*  degv  = (float*)alloc((size_t)N * 4);
  float*  smalls = (float*)alloc(512 * 4);
  float *u_cur = smalls, *uD = smalls + 32, *hbuf = smalls + 64, *cbuf = smalls + 96;
  float *qbuf = smalls + 128, *nvec = smalls + 192, *evec = smalls + 256, *gm = smalls + 320;
  float*  logits = (float*)alloc((size_t)E * 4);
  float*  pmax = (float*)alloc(G_RED * 4);
  float*  psum = (float*)alloc(G_RED * 4);
  float*  pr   = (float*)alloc((size_t)G_RED * 32 * 4);
  int*    focus = (int*)alloc(256);

  auto gs = [](long n, int t) { return (unsigned)((n + t - 1) / t); };
  const unsigned EB = gs(E, 64), NB = gs(N, 64);

  // --- setup: degree, focus, encoders ---
  mg_fill<<<gs(N, 256), 256, 0, stream>>>(degv, N, 0.0f);
  mg_deg<<<1024, 256, 0, stream>>>(dstp, degv, E);
  mg_focus<<<gs(N, 256), 256, 0, stream>>>(node_feat, focus, N);
  mg_node_init<<<NB, 128, 0, stream>>>(node_feat, node_emb, neW0, neB0, neW1, neB1, v_cur, N);
  mg_edge_init<<<EB, 128, 0, stream>>>(edge_dist, eeW0, eeB0, eeW1, eeB1, e_cur, E);
  mg_state_enc<<<1, 128, 0, stream>>>(state_feat, state_emb, focus, scW0, scB0, scW1, scB1, u_cur);

  // --- 3 MEGNet blocks ---
  for (int b = 0; b < 3; ++b) {
    if (b > 0) {
      mg_dense_rows<<<EB, 128, 0, stream>>>(e_cur, deW0[b], deB0[b], deW1[b], deB1[b], eD, E);
      mg_dense_rows<<<NB, 128, 0, stream>>>(v_cur, dnW0[b], dnB0[b], dnW1[b], dnB1[b], vD, N);
      mg_dense_state<<<1, 128, 0, stream>>>(u_cur, dsW0[b], dsB0[b], dsW1[b], dsB1[b], uD);
    } else {
      mg_cvt<<<2048, 256, 0, stream>>>(e_cur, eD, (long)E * 32);
      mg_cvt<<<1024, 256, 0, stream>>>(v_cur, vD, (long)N * 32);
      mg_copyf<<<1, 32, 0, stream>>>(u_cur, uD, 32);
    }
    mg_fill<<<gs((long)N * 32 + 64, 256), 256, 0, stream>>>(e_sum, (long)N * 32 + 64, 0.0f);
    mg_edge_update<<<EB, 128, 0, stream>>>(srcp, dstp, eD, vD, uD,
                                           edW[b][0], edB[b][0], edW[b][1], edB[b][1], edW[b][2], edB[b][2],
                                           e_cur, e_sum, eg_sum, E);
    mg_node_update<<<NB, 128, 0, stream>>>(vD, e_sum, degv, uD,
                                           ndW[b][0], ndB[b][0], ndW[b][1], ndB[b][1], ndW[b][2], ndB[b][2],
                                           v_cur, vg_sum, N);
    mg_state_update<<<1, 128, 0, stream>>>(eg_sum, vg_sum, uD,
                                           stW[b][0], stB[b][0], stW[b][1], stB[b][1], stW[b][2], stB[b][2],
                                           u_cur, 1.0f / (float)E, 1.0f / (float)N);
  }

  // --- set2set readouts ---
  auto run_s2s = [&](const float* feat, int M, const float* Wih, const float* Whh, const float* bb, float* outv) {
    mg_fill<<<1, 128, 0, stream>>>(hbuf, 128, 0.0f);   // h, c, q contiguous
    for (int it = 0; it < 2; ++it) {
      mg_lstm<<<1, 128, 0, stream>>>(qbuf, hbuf, cbuf, Wih, Whh, bb);
      mg_logits<<<G_RED, 256, 0, stream>>>(feat, M, hbuf, logits, pmax);
      mg_maxred<<<1, 256, 0, stream>>>(pmax, G_RED, gm);
      mg_expsum<<<G_RED, 256, 0, stream>>>(feat, M, logits, gm, psum, pr);
      mg_s2s_final<<<1, 256, 0, stream>>>(psum, pr, G_RED, hbuf, qbuf, outv);
    }
  };
  run_s2s(v_cur, N, snWih, snWhh, snB, nvec);
  run_s2s(e_cur, E, szWih, szWhh, szB, evec);

  // --- output MLP ---
  mg_out<<<1, 128, 0, stream>>>(nvec, evec, u_cur, oW0, oB0, oW1, oB1, oW2, oB2, (float*)d_out);
}
